// MultiHeadAttention_21663815041816
// MI455X (gfx1250) — matrix operand order repaired and checked by reading, not compiled, not run on hardware
//
#include <hip/hip_runtime.h>
#include <hip/hip_bf16.h>

// ---------------------------------------------------------------------------
// MultiHeadAttention forward, CDNA5 (gfx1250), wave32.
// bf16 WMMA (f32 accum) + async global->LDS weight staging + flash attention.
// B=8, T=1024, E=1024, H=16, HS=64.
// ---------------------------------------------------------------------------

typedef __bf16 bf16_t;
typedef __attribute__((ext_vector_type(16))) __bf16 v16bf;
typedef __attribute__((ext_vector_type(8)))  __bf16 v8bf;
typedef __attribute__((ext_vector_type(4)))  __bf16 v4bf;
typedef __attribute__((ext_vector_type(8)))  float  v8f;

constexpr int Bc  = 8;
constexpr int Tc  = 1024;
constexpr int Ec  = 1024;
constexpr int Hc  = 16;
constexpr int HSc = 64;

// Weight tile staged in LDS: 64 N-rows x 32 K, rows padded to 40 bf16 (80 B)
// so the 16-lane b128 fragment reads land on 16 distinct banks.
constexpr int WT_PAD = 40;

__device__ __forceinline__ bf16_t f2bf(float x) {
    union { float f; unsigned u; } v; v.f = x;
    unsigned r = v.u + 0x7FFFu + ((v.u >> 16) & 1u);   // round-to-nearest-even
    unsigned short h = (unsigned short)(r >> 16);
    return __builtin_bit_cast(bf16_t, h);
}

// ---- Fragment loaders (16-bit element layouts per CDNA5 ISA 7.12.2) -------
// A-matrix 16x32 (MxK): lane L<16 -> row M=L, elems 0..7 = K=k0..k0+7,
// elems 8..15 = K=k0+16..k0+23; lanes 16..31 same rows with K offset +8.
__device__ __forceinline__ v16bf load_a_frag(const bf16_t* __restrict__ base,
                                             int ld, int row0, int k0) {
    const int lane = threadIdx.x & 31;
    const int half = lane >> 4;
    const int r    = lane & 15;
    const bf16_t* p = base + (size_t)(row0 + r) * ld + (k0 + half * 8);
    v8bf lo = *reinterpret_cast<const v8bf*>(p);
    v8bf hi = *reinterpret_cast<const v8bf*>(p + 16);
    v16bf f;
#pragma unroll
    for (int i = 0; i < 8; ++i) { f[i] = lo[i]; f[i + 8] = hi[i]; }
    return f;
}

// B-matrix 32x16 (KxN), B[K][N] = W[N][K] row-major:
// lanes 0..15 hold K=k0..k0+7, k0+16..k0+23 for N=lane; lanes 16..31 same with K offset +8.
__device__ __forceinline__ v16bf load_b_frag(const bf16_t* __restrict__ base,
                                             int ld, int n0, int k0) {
    const int lane = threadIdx.x & 31;
    const int half = lane >> 4;
    const int c    = lane & 15;
    const bf16_t* p = base + (size_t)(n0 + c) * ld + (k0 + half * 8);
    v8bf lo = *reinterpret_cast<const v8bf*>(p);
    v8bf hi = *reinterpret_cast<const v8bf*>(p + 16);
    v16bf f;
#pragma unroll
    for (int i = 0; i < 8; ++i) { f[i] = lo[i]; f[i + 8] = hi[i]; }
    return f;
}

// Same B-fragment layout, sourced from the LDS-staged [64][WT_PAD] tile.
__device__ __forceinline__ v16bf load_b_frag_lds(const bf16_t* tile, int n0j) {
    const int lane = threadIdx.x & 31;
    const int half = lane >> 4;
    const int c    = lane & 15;
    const bf16_t* p = tile + (size_t)(n0j + c) * WT_PAD + half * 8;
    v8bf lo = *reinterpret_cast<const v8bf*>(p);
    v8bf hi = *reinterpret_cast<const v8bf*>(p + 16);
    v16bf f;
#pragma unroll
    for (int i = 0; i < 8; ++i) { f[i] = lo[i]; f[i + 8] = hi[i]; }
    return f;
}

__device__ __forceinline__ v8f wmma_bf16(v16bf a, v16bf b, v8f c) {
    return __builtin_amdgcn_wmma_f32_16x16x32_bf16(
        /*neg_a=*/false, a, /*neg_b=*/false, b,
        /*c_mod=*/(short)0, c, /*reuse_a=*/false, /*reuse_b=*/false);
}

// Async global -> LDS copy of 16 bytes per lane (CDNA5, tracked by ASYNCcnt).
// LDS address = low 32 bits of the shared-aperture flat pointer (ISA 10.2).
__device__ __forceinline__ void async_copy_b128(const bf16_t* gptr, void* lptr) {
    const unsigned ldsa = (unsigned)(uintptr_t)lptr;
    asm volatile("global_load_async_to_lds_b128 %0, %1, off"
                 :: "v"(ldsa), "v"(gptr) : "memory");
}
__device__ __forceinline__ void wait_async0() {
    asm volatile("s_wait_asynccnt 0x0" ::: "memory");
}

// Reductions across the 16 lanes of one wave-half (rows are per-half in the
// C/D layout: lane<16 vgpr r -> row r, lane>=16 vgpr r -> row 8+r).
__device__ __forceinline__ float half_max(float v) {
#pragma unroll
    for (int m = 1; m <= 8; m <<= 1) v = fmaxf(v, __shfl_xor(v, m, 32));
    return v;
}
__device__ __forceinline__ float half_sum(float v) {
#pragma unroll
    for (int m = 1; m <= 8; m <<= 1) v += __shfl_xor(v, m, 32);
    return v;
}

// ---------------------------------------------------------------------------
// f32 -> bf16 conversion, 4 elements per thread.
// ---------------------------------------------------------------------------
__global__ void cvt_f32_bf16(const float* __restrict__ src,
                             bf16_t* __restrict__ dst, int n4) {
    int i = blockIdx.x * blockDim.x + threadIdx.x;
    if (i < n4) {
        float4 v = reinterpret_cast<const float4*>(src)[i];
        v4bf o;
        o[0] = f2bf(v.x); o[1] = f2bf(v.y); o[2] = f2bf(v.z); o[3] = f2bf(v.w);
        reinterpret_cast<v4bf*>(dst)[i] = o;
    }
}

// ---------------------------------------------------------------------------
// GEMM: out[m, n] = sum_k A[m,k] * W[n,k], per batch b.
// Block = 8 waves = 128(M) x 64(N); the shared 64x32 W tile is staged into
// LDS with async b128 copies (double buffered), A fragments come straight
// from global (unique per wave).
// MODE 0: bf16 out [B,H,T,HS] (q,k)   MODE 1: bf16 out [B,H,HS,T] (v, transp)
// MODE 2: f32 out [B,T,E] (final projection)
// ---------------------------------------------------------------------------
template <int MODE>
__global__ void gemm_wmma(const bf16_t* __restrict__ ab,
                          const bf16_t* __restrict__ wb,
                          void* __restrict__ out_) {
    __shared__ __align__(16) bf16_t wtile[2][64][WT_PAD];

    const int tid  = threadIdx.x;
    const int wave = tid >> 5;
    const int lane = tid & 31;
    const int half = lane >> 4;
    const int c    = lane & 15;
    const int n0   = blockIdx.x * 64;
    const int m0   = blockIdx.y * 128 + wave * 16;
    const int b    = blockIdx.z;
    const bf16_t* A = ab + (size_t)b * Tc * Ec;

    // Cooperative async fill: 256 threads x 16 B = the full 64x32 bf16 tile.
    const int fn = tid >> 2;         // 0..63  : W row within tile
    const int fk = (tid & 3) * 8;    // 0,8,16,24 : K element offset
    auto fill = [&](int buf, int k0) {
        async_copy_b128(wb + (size_t)(n0 + fn) * Ec + (k0 + fk),
                        &wtile[buf][fn][fk]);
    };

    fill(0, 0);
    v8f acc[4] = {};
    for (int k0 = 0; k0 < Ec; k0 += 32) {
        const int buf = (k0 >> 5) & 1;
        wait_async0();                 // own fills of `buf` have landed
        __syncthreads();               // everyone's fills landed; everyone is
                                       // done reading buf^1 from last step
        if (k0 + 32 < Ec) fill(buf ^ 1, k0 + 32);
        v16bf a = load_a_frag(A, Ec, m0, k0);
#pragma unroll
        for (int j = 0; j < 4; ++j) {
            v16bf bf = load_b_frag_lds(&wtile[buf][0][0], j * 16);
            acc[j] = wmma_bf16(a, bf, acc[j]);
        }
    }

    if (MODE == 2) {
        float* out = (float*)out_;
#pragma unroll
        for (int j = 0; j < 4; ++j)
#pragma unroll
            for (int r = 0; r < 8; ++r) {
                const int t = m0 + half * 8 + r;
                out[((size_t)b * Tc + t) * Ec + (n0 + j * 16 + c)] = acc[j][r];
            }
    } else {
        bf16_t* out = (bf16_t*)out_;
        const int h = n0 >> 6;
#pragma unroll
        for (int j = 0; j < 4; ++j) {
            const int d = j * 16 + c;
#pragma unroll
            for (int r = 0; r < 8; ++r) {
                const int t = m0 + half * 8 + r;
                const bf16_t val = f2bf(acc[j][r]);
                if (MODE == 0)
                    out[(((size_t)b * Hc + h) * Tc + t) * HSc + d] = val;
                else
                    out[(((size_t)b * Hc + h) * HSc + d) * Tc + t] = val;
            }
        }
    }
}

// ---------------------------------------------------------------------------
// Causal flash attention. One wave owns 16 query rows of one (b,h); streams
// keys in 32-wide chunks with online softmax. P transposed through LDS
// (C-layout store -> A-layout load).  attn out: bf16 [B, T, E] (heads concat).
// ---------------------------------------------------------------------------
__global__ void flash_attn(const bf16_t* __restrict__ qb,   // [B,H,T,HS]
                           const bf16_t* __restrict__ kb,   // [B,H,T,HS]
                           const bf16_t* __restrict__ vtb,  // [B,H,HS,T]
                           bf16_t* __restrict__ attnb) {    // [B,T,E]
    const int wave = threadIdx.x >> 5;
    const int lane = threadIdx.x & 31;
    const int half = lane >> 4;
    const int c    = lane & 15;
    const int h    = blockIdx.y;
    const int b    = blockIdx.z;
    const int t0   = blockIdx.x * 128 + wave * 16;

    const bf16_t* Q  = qb  + ((size_t)b * Hc + h) * Tc * HSc;
    const bf16_t* K  = kb  + ((size_t)b * Hc + h) * Tc * HSc;
    const bf16_t* VT = vtb + ((size_t)b * Hc + h) * HSc * Tc;

    __shared__ __align__(16) bf16_t pbuf[8][16][32];

    // Query fragments (K-dim = HS = 64 -> two 32-deep fragments), reused all loop.
    const v16bf aq0 = load_a_frag(Q, HSc, t0, 0);
    const v16bf aq1 = load_a_frag(Q, HSc, t0, 32);

    v8f o[4] = {};
    float mrow[8], lrow[8];
#pragma unroll
    for (int r = 0; r < 8; ++r) { mrow[r] = -1e30f; lrow[r] = 0.f; }

    const float sc = 0.125f; // HS^-1/2

    for (int s0 = 0; s0 < t0 + 16; s0 += 32) {           // wave-uniform bound
        // S tile a: keys s0..s0+15
        v8f sa = {};
        {
            v16bf bk0 = load_b_frag(K + (size_t)s0 * HSc, HSc, 0, 0);
            v16bf bk1 = load_b_frag(K + (size_t)s0 * HSc, HSc, 0, 32);
            sa = wmma_bf16(aq0, bk0, sa);
            sa = wmma_bf16(aq1, bk1, sa);
        }
        // S tile b: keys s0+16..s0+31 (skip if fully above the diagonal)
        const bool do_b = (s0 + 16) < (t0 + 16);          // wave-uniform
        v8f sb = {};
        if (do_b) {
            v16bf bk0 = load_b_frag(K + (size_t)(s0 + 16) * HSc, HSc, 0, 0);
            v16bf bk1 = load_b_frag(K + (size_t)(s0 + 16) * HSc, HSc, 0, 32);
            sb = wmma_bf16(aq0, bk0, sb);
            sb = wmma_bf16(aq1, bk1, sb);
        }
        // Online softmax per row (rows live per wave-half).
#pragma unroll
        for (int r = 0; r < 8; ++r) {
            const int row = t0 + half * 8 + r;
            const float va = (s0 + c      <= row) ? sa[r] * sc : -1e30f;
            const float vb = (do_b && (s0 + 16 + c <= row)) ? sb[r] * sc : -1e30f;
            const float tmax = half_max(fmaxf(va, vb));
            const float mnew = fmaxf(mrow[r], tmax);
            const float resc = __expf(mrow[r] - mnew);
            const float pa = __expf(va - mnew);
            const float pb = __expf(vb - mnew);
            lrow[r] = lrow[r] * resc + half_sum(pa + pb);
            mrow[r] = mnew;
#pragma unroll
            for (int j = 0; j < 4; ++j) o[j][r] *= resc;
            pbuf[wave][half * 8 + r][c]      = f2bf(pa);
            pbuf[wave][half * 8 + r][16 + c] = f2bf(pb);
        }
        asm volatile("s_wait_dscnt 0x0" ::: "memory");    // LDS store->load, same wave
        // Reload P in A-fragment layout (LDS did the lane transpose).
        v16bf pf;
        {
            const bf16_t* pr = &pbuf[wave][c][half * 8];
            v8bf lo = *reinterpret_cast<const v8bf*>(pr);
            v8bf hi = *reinterpret_cast<const v8bf*>(pr + 16);
#pragma unroll
            for (int i = 0; i < 8; ++i) { pf[i] = lo[i]; pf[i + 8] = hi[i]; }
        }
        // O += P (16x32) x V (32x64): B-frags contiguous thanks to v-transpose.
#pragma unroll
        for (int j = 0; j < 4; ++j) {
            v16bf vf = load_b_frag(VT, Tc, j * 16, s0);
            o[j] = wmma_bf16(pf, vf, o[j]);
        }
    }

    // Normalize and store heads-concatenated [B,T,E] in bf16.
#pragma unroll
    for (int r = 0; r < 8; ++r) {
        const int t = t0 + half * 8 + r;
        const float inv = 1.f / lrow[r];
#pragma unroll
        for (int j = 0; j < 4; ++j) {
            attnb[((size_t)b * Tc + t) * Ec + (h * HSc + j * 16 + c)] =
                f2bf(o[j][r] * inv);
        }
    }
}

// ---------------------------------------------------------------------------
// Host launch
// ---------------------------------------------------------------------------
extern "C" void kernel_launch(void* const* d_in, const int* in_sizes, int n_in,
                              void* d_out, int out_size, void* d_ws, size_t ws_size,
                              hipStream_t stream) {
    const float* x  = (const float*)d_in[0];
    const float* Wk = (const float*)d_in[1];
    const float* Wq = (const float*)d_in[2];
    const float* Wv = (const float*)d_in[3];
    const float* Wp = (const float*)d_in[4];
    float* out = (float*)d_out;

    const size_t NX = (size_t)Bc * Tc * Ec;       // 8M
    const size_t NW = (size_t)Ec * Ec;            // 1M
    const size_t NQ = (size_t)Bc * Hc * Tc * HSc; // 8M

    char* ws = (char*)d_ws;
    size_t off = 0;
    auto carve = [&](size_t elems) {
        bf16_t* p = (bf16_t*)(ws + off);
        off += ((elems * sizeof(bf16_t)) + 255) & ~(size_t)255;
        return p;
    };
    bf16_t* xb    = carve(NX);
    bf16_t* wqb   = carve(NW);
    bf16_t* wkb   = carve(NW);
    bf16_t* wvb   = carve(NW);
    bf16_t* wpb   = carve(NW);
    bf16_t* qbuf  = carve(NQ);
    bf16_t* kbuf  = carve(NQ);
    bf16_t* vtbuf = carve(NQ);
    bf16_t* attnb = carve(NX);
    (void)ws_size; (void)in_sizes; (void)n_in; (void)out_size;

    auto launch_cvt = [&](const float* s, bf16_t* d, size_t n) {
        int n4 = (int)(n / 4);
        cvt_f32_bf16<<<(n4 + 255) / 256, 256, 0, stream>>>(s, d, n4);
    };
    launch_cvt(x,  xb,  NX);
    launch_cvt(Wq, wqb, NW);
    launch_cvt(Wk, wkb, NW);
    launch_cvt(Wv, wvb, NW);
    launch_cvt(Wp, wpb, NW);

    dim3 gGemm(Ec / 64, Tc / 128, Bc);   // (16, 8, 8)
    gemm_wmma<0><<<gGemm, 256, 0, stream>>>(xb, wqb, qbuf);
    gemm_wmma<0><<<gGemm, 256, 0, stream>>>(xb, wkb, kbuf);
    gemm_wmma<1><<<gGemm, 256, 0, stream>>>(xb, wvb, vtbuf);

    dim3 gAttn(Tc / 128, Hc, Bc);        // (8, 16, 8)
    flash_attn<<<gAttn, 256, 0, stream>>>(qbuf, kbuf, vtbuf, attnb);

    gemm_wmma<2><<<gGemm, 256, 0, stream>>>(attnb, wpb, out);
}
